// Recurrent_26439818674415
// MI455X (gfx1250) — matrix operand order repaired and checked by reading, not compiled, not run on hardware
//
#include <hip/hip_runtime.h>
#include <stdint.h>

// ---------------- problem constants ----------------
#define B_   64
#define T_   1024
#define H_   512
#define L_   2
#define G4_  2048               // 4*H
#define BH   (B_*H_)            // 32768
#define NBLK 32                 // H/16 column slices -> one block each
#define THREADS 256             // 8 wave32 waves

typedef __bf16 bf16_t;
typedef __attribute__((ext_vector_type(16))) __bf16 bf16x16;
typedef __attribute__((ext_vector_type(8)))  float  floatx8;

union AF { bf16x16 v; uint4 q[2]; };   // 32B fragment: two 16B global loads

// ---------------- workspace layout (bytes) ----------------
#define WS_CNT   0                                  // 2 uints (grid barrier), pad 1KB
#define WS_XBUF  1024                               // bf16 [B][H]   (x_t for layer 0)
#define WS_HBUF  (WS_XBUF + BH*2)                   // bf16 [L][2][B][H] double-buffered h
#define WS_WIH   (WS_HBUF + L_*2*BH*2)              // bf16 [L][4H][H]
#define WS_WHH   (WS_WIH + (size_t)L_*G4_*H_*2)     // bf16 [L][4H][H]

#define OUT_HT   ((size_t)B_*T_*H_)                 // 33,554,432 floats
#define OUT_CT   (OUT_HT + (size_t)L_*BH)

// ---------------- helpers ----------------
__device__ __forceinline__ float sigf(float x)  { return 1.0f / (1.0f + __expf(-x)); }
__device__ __forceinline__ float tanh_(float x) { return 1.0f - 2.0f / (__expf(2.0f*x) + 1.0f); }
__device__ __forceinline__ bf16_t f2bf(float f) { return (bf16_t)f; }

__device__ __forceinline__ void grid_barrier(unsigned* cnt, unsigned* gen, unsigned nblk) {
    __syncthreads();
    if (threadIdx.x == 0) {
        __threadfence();  // release prior global writes (device scope)
        unsigned g = __hip_atomic_load(gen, __ATOMIC_RELAXED, __HIP_MEMORY_SCOPE_AGENT);
        unsigned a = __hip_atomic_fetch_add(cnt, 1u, __ATOMIC_ACQ_REL, __HIP_MEMORY_SCOPE_AGENT);
        if (a == nblk - 1u) {
            __hip_atomic_store(cnt, 0u, __ATOMIC_RELAXED, __HIP_MEMORY_SCOPE_AGENT);
            __hip_atomic_store(gen, g + 1u, __ATOMIC_RELEASE, __HIP_MEMORY_SCOPE_AGENT);
        } else {
            while (__hip_atomic_load(gen, __ATOMIC_ACQUIRE, __HIP_MEMORY_SCOPE_AGENT) == g) {
                __builtin_amdgcn_s_sleep(1);
            }
        }
        __threadfence();  // acquire
    }
    __syncthreads();
}

// ---------------- prep: f32 -> bf16 weights / h0 / x(t=0) ----------------
__global__ void lstm_prep(const float* __restrict__ Wih, const float* __restrict__ Whh,
                          const float* __restrict__ h0,  const float* __restrict__ inputs,
                          bf16_t* __restrict__ wbih, bf16_t* __restrict__ wbhh,
                          bf16_t* __restrict__ hbuf, bf16_t* __restrict__ xbuf) {
    size_t i = (size_t)blockIdx.x * blockDim.x + threadIdx.x;
    const size_t NW = (size_t)L_ * G4_ * H_;   // 2,097,152 per tensor
    if (i < NW) { wbih[i] = f2bf(Wih[i]); wbhh[i] = f2bf(Whh[i]); }
    if (i < (size_t)L_*BH) {
        size_t l = i / BH, r = i % BH;
        hbuf[(l*2 + 0)*BH + r] = f2bf(h0[i]);   // parity 0 holds initial h
    }
    if (i < (size_t)BH) {
        size_t b = i >> 9, n = i & 511;
        xbuf[i] = f2bf(inputs[b*(size_t)T_*H_ + n]);   // x at t=0
    }
}

// ---------------- one layer-step for this block's 16-column H slice ----------------
// Wave mapping: wave = gate*? no -> gate = wave&3 (i,f,g,o), s = wave>>2
//   s==0: stream A = x,  B = W_ih ;  s==1: stream A = h_prev, B = W_hh
// Each wave computes all 4 M-tiles for its (gate, stream): B fragment loaded once
// per k-step, reused across 4 WMMAs. Partials summed in LDS planes Gs[s].
__device__ __forceinline__ void do_layer(
    int l, int t, int blk, int tid,
    const bf16_t* __restrict__ xsrc,   // [B][H] bf16 (layer input)
    const bf16_t* __restrict__ hsrc,   // [B][H] bf16 (previous h, this layer)
    bf16_t* __restrict__ hdst,         // [B][H] bf16 (new h, this layer)
    const bf16_t* __restrict__ wih,    // [4H][H] bf16
    const bf16_t* __restrict__ whh,    // [4H][H] bf16
    float (*Gs)[64][65], float (*Csl)[16], const float* __restrict__ Bsl,
    float* __restrict__ out, bool top, bool last)
{
    const int wave = tid >> 5, lane = tid & 31;
    const int gate = wave & 3;           // 0..3 : i,f,g,o
    const int s    = wave >> 2;          // 0: x*Wih, 1: h*Whh
    const int hi   = lane >> 4, lo = lane & 15;

    const bf16_t* asrc = s ? hsrc : xsrc;
    const bf16_t* wsrc = s ? whh  : wih;

    const int wrow = gate*H_ + blk*16 + lo;        // weight row = gate output index
    const bf16_t* wp = wsrc + (size_t)wrow * H_;
    const bf16_t* a0 = asrc + (size_t)(lo     ) * H_;   // M-tile 0 row
    const bf16_t* a1 = asrc + (size_t)(lo + 16) * H_;   // M-tile 1 row
    const bf16_t* a2 = asrc + (size_t)(lo + 32) * H_;   // M-tile 2 row
    const bf16_t* a3 = asrc + (size_t)(lo + 48) * H_;   // M-tile 3 row

    floatx8 c0 = {}, c1 = {}, c2 = {}, c3 = {};

    #pragma unroll 2
    for (int kk = 0; kk < H_; kk += 32) {
        const int ka = kk + hi*8;    // A lane chunk base (8 elems + 8 elems @ +16)
        const int kb = kk + hi*8;    // B lane chunk base (8 elems + 8 elems @ +16)
        AF bw, f0, f1, f2, f3;
        bw.q[0] = *(const uint4*)(wp + kb);  bw.q[1] = *(const uint4*)(wp + kb + 16);
        f0.q[0] = *(const uint4*)(a0 + ka);  f0.q[1] = *(const uint4*)(a0 + ka + 16);
        f1.q[0] = *(const uint4*)(a1 + ka);  f1.q[1] = *(const uint4*)(a1 + ka + 16);
        f2.q[0] = *(const uint4*)(a2 + ka);  f2.q[1] = *(const uint4*)(a2 + ka + 16);
        f3.q[0] = *(const uint4*)(a3 + ka);  f3.q[1] = *(const uint4*)(a3 + ka + 16);
        c0 = __builtin_amdgcn_wmma_f32_16x16x32_bf16(false, f0.v, false, bw.v, (short)0, c0, false, false);
        c1 = __builtin_amdgcn_wmma_f32_16x16x32_bf16(false, f1.v, false, bw.v, (short)0, c1, false, false);
        c2 = __builtin_amdgcn_wmma_f32_16x16x32_bf16(false, f2.v, false, bw.v, (short)0, c2, false, false);
        c3 = __builtin_amdgcn_wmma_f32_16x16x32_bf16(false, f3.v, false, bw.v, (short)0, c3, false, false);
    }

    // C/D layout: VGPR r -> M = r + 8*hi, N = lo ; stage per-stream partials
    #pragma unroll
    for (int r = 0; r < 8; ++r) {
        const int mr = r + hi*8;
        Gs[s][     mr][gate*16 + lo] = c0[r];
        Gs[s][16 + mr][gate*16 + lo] = c1[r];
        Gs[s][32 + mr][gate*16 + lo] = c2[r];
        Gs[s][48 + mr][gate*16 + lo] = c3[r];
    }
    __syncthreads();

    // elementwise LSTM cell: 64 batch x 16 cols = 1024 elems / 256 threads
    #pragma unroll
    for (int e = tid; e < 64*16; e += THREADS) {
        const int b = e >> 4, nc = e & 15;
        const float gi = Gs[0][b][     nc] + Gs[1][b][     nc] + Bsl[     nc];
        const float gf = Gs[0][b][16 + nc] + Gs[1][b][16 + nc] + Bsl[16 + nc];
        const float gg = Gs[0][b][32 + nc] + Gs[1][b][32 + nc] + Bsl[32 + nc];
        const float go = Gs[0][b][48 + nc] + Gs[1][b][48 + nc] + Bsl[48 + nc];
        const float co = Csl[b][nc];
        const float cn = sigf(gf) * co + sigf(gi) * tanh_(gg);
        const float hn = sigf(go) * tanh_(cn);
        Csl[b][nc] = cn;
        const int n = blk*16 + nc;
        hdst[(size_t)b*H_ + n] = f2bf(hn);
        if (top)  out[(size_t)b*T_*H_ + (size_t)t*H_ + n] = hn;
        if (last) {
            out[OUT_HT + (size_t)l*BH + (size_t)b*H_ + n] = hn;
            out[OUT_CT + (size_t)l*BH + (size_t)b*H_ + n] = cn;
        }
    }
}

// ---------------- persistent recurrent kernel ----------------
__launch_bounds__(THREADS, 1)
__global__ void lstm_persist(const float* __restrict__ inputs, const float* __restrict__ c0,
                             const float* __restrict__ b_ih,  const float* __restrict__ b_hh,
                             float* __restrict__ out,
                             bf16_t* __restrict__ xbuf, bf16_t* __restrict__ hbuf,
                             const bf16_t* __restrict__ wbih, const bf16_t* __restrict__ wbhh,
                             unsigned* __restrict__ bar)
{
    __shared__ float Gs[2][64][65];     // per-stream gate partials, padded
    __shared__ float Cs[L_][64][16];    // block-resident cell state, both layers
    __shared__ float Bs[L_][64];        // fused biases for this column slice

    const int tid = threadIdx.x;
    const int blk = blockIdx.x;         // H columns [blk*16, blk*16+16)

    // init block-local state (c0, biases)
    for (int i = tid; i < L_*64*16; i += THREADS) {
        const int l = i >> 10, r = i & 1023, b = r >> 4, nc = r & 15;
        Cs[l][b][nc] = c0[(size_t)l*BH + (size_t)b*H_ + blk*16 + nc];
    }
    for (int i = tid; i < L_*64; i += THREADS) {
        const int l = i >> 6, j = i & 63, g = j >> 4, nc = j & 15;
        const int col = g*H_ + blk*16 + nc;
        Bs[l][j] = b_ih[l*G4_ + col] + b_hh[l*G4_ + col];
    }
    __syncthreads();

    unsigned* cnt = bar;
    unsigned* gen = bar + 1;

    for (int t = 0; t < T_; ++t) {
        const int rp = t & 1, wp = rp ^ 1;

        // ---- layer 0: x = xbuf (bf16 of inputs[:,t,:]) ----
        do_layer(0, t, blk, tid,
                 xbuf, hbuf + (size_t)(0*2 + rp)*BH, hbuf + (size_t)(0*2 + wp)*BH,
                 wbih + (size_t)0*G4_*H_, wbhh + (size_t)0*G4_*H_,
                 Gs, Cs[0], Bs[0], out, false, t == T_-1);
        grid_barrier(cnt, gen, NBLK);

        // ---- layer 1: x = new h of layer 0 ----
        do_layer(1, t, blk, tid,
                 hbuf + (size_t)(0*2 + wp)*BH, hbuf + (size_t)(1*2 + rp)*BH, hbuf + (size_t)(1*2 + wp)*BH,
                 wbih + (size_t)1*G4_*H_, wbhh + (size_t)1*G4_*H_,
                 Gs, Cs[1], Bs[1], out, true, t == T_-1);

        // overlap: stage x_{t+1} -> bf16 xbuf (readers of x_t finished before last barrier)
        if (t + 1 < T_) {
            for (int i = blk*THREADS + tid; i < BH; i += NBLK*THREADS) {
                const int b = i >> 9, n = i & 511;
                xbuf[i] = f2bf(inputs[(size_t)b*T_*H_ + (size_t)(t+1)*H_ + n]);
            }
        }
        grid_barrier(cnt, gen, NBLK);
    }
}

// ---------------- launch ----------------
extern "C" void kernel_launch(void* const* d_in, const int* in_sizes, int n_in,
                              void* d_out, int out_size, void* d_ws, size_t ws_size,
                              hipStream_t stream) {
    const float* inputs = (const float*)d_in[0];
    const float* h0     = (const float*)d_in[1];
    const float* c0     = (const float*)d_in[2];
    const float* W_ih   = (const float*)d_in[3];
    const float* W_hh   = (const float*)d_in[4];
    const float* b_ih   = (const float*)d_in[5];
    const float* b_hh   = (const float*)d_in[6];
    float* out = (float*)d_out;

    char* ws = (char*)d_ws;
    unsigned* bar  = (unsigned*)(ws + WS_CNT);
    bf16_t*  xbuf  = (bf16_t*)(ws + WS_XBUF);
    bf16_t*  hbuf  = (bf16_t*)(ws + WS_HBUF);
    bf16_t*  wbih  = (bf16_t*)(ws + WS_WIH);
    bf16_t*  wbhh  = (bf16_t*)(ws + WS_WHH);

    // zero grid-barrier state (capture-legal)
    hipMemsetAsync(d_ws, 0, 1024, stream);

    // one-shot f32->bf16 conversion of weights / h0 / x(t=0)
    const int prep_threads = 256;
    const int prep_blocks  = (int)(((size_t)L_*G4_*H_ + prep_threads - 1) / prep_threads); // 8192
    lstm_prep<<<prep_blocks, prep_threads, 0, stream>>>(W_ih, W_hh, h0, inputs, wbih, wbhh, hbuf, xbuf);

    // persistent recurrent kernel: 32 resident blocks, device-wide barriers per layer-step
    lstm_persist<<<NBLK, THREADS, 0, stream>>>(inputs, c0, b_ih, b_hh, out,
                                               xbuf, hbuf, wbih, wbhh, bar);
}
